// Block_32521492365607
// MI455X (gfx1250) — matrix operand order repaired and checked by reading, not compiled, not run on hardware
//
#include <hip/hip_runtime.h>
#include <hip/hip_bf16.h>

typedef __attribute__((ext_vector_type(16))) __bf16 v16bf;
typedef __attribute__((ext_vector_type(8)))  __bf16 v8bf;
typedef __attribute__((ext_vector_type(8)))  float  v8f;
typedef int v4i __attribute__((vector_size(16)));   // matches builtin's V4i

#define ASPACE(n) __attribute__((address_space(n)))

#define WMMA_BF16(a, b, c) \
  __builtin_amdgcn_wmma_f32_16x16x32_bf16(false, (a), false, (b), (short)0, (c), false, false)

// ---------------- async global -> LDS copy (CDNA5 GLOBAL_LOAD_ASYNC_TO_LDS_B128) ----
static __device__ inline void async_copy_b128(const __bf16* gptr, __bf16* lptr) {
#if __has_builtin(__builtin_amdgcn_global_load_async_to_lds_b128)
  __builtin_amdgcn_global_load_async_to_lds_b128(
      (ASPACE(1) v4i*)(uintptr_t)gptr,
      (ASPACE(3) v4i*)(unsigned)(uintptr_t)lptr, 0, 0);
#else
  unsigned lds = (unsigned)(uintptr_t)lptr;
  unsigned long long ga = (unsigned long long)(uintptr_t)gptr;
  asm volatile("global_load_async_to_lds_b128 %0, %1, off"
               :: "v"(lds), "v"(ga) : "memory");
#endif
}
static __device__ inline void wait_async0() {
#if __has_builtin(__builtin_amdgcn_s_wait_asynccnt)
  __builtin_amdgcn_s_wait_asynccnt(0);
#else
  asm volatile("s_wait_asynccnt 0x0" ::: "memory");
#endif
}

// ---------------- fragment load helpers ----------------
// A-matrix 16x32 bf16 layout: lane holds row r = lane&15; element j -> K = 16*hi + j.
static __device__ inline v16bf ld_fragA(const __bf16* row, int hi) {
  v8bf lo = *(const v8bf*)(row + 16 * hi);
  v8bf hh = *(const v8bf*)(row + 16 * hi + 8);
  return __builtin_shufflevector(lo, hh, 0,1,2,3,4,5,6,7,8,9,10,11,12,13,14,15);
}
// B-matrix 32x16 bf16 layout: lane holds column n = lane&15; element j -> K = 16*hi + j.
// Source stored as B^T [N][K] row-major -> contiguous per-lane chunk.
static __device__ inline v16bf ld_fragB(const __bf16* row, int hi) {
  v8bf lo = *(const v8bf*)(row + 16 * hi);
  v8bf hh = *(const v8bf*)(row + 16 * hi + 8);
  return __builtin_shufflevector(lo, hh, 0,1,2,3,4,5,6,7,8,9,10,11,12,13,14,15);
}

static __device__ inline float hmax16(float x) {
  for (int m = 1; m < 16; m <<= 1) x = fmaxf(x, __shfl_xor(x, m, 32));
  return x;
}
static __device__ inline float hsum16(float x) {
  for (int m = 1; m < 16; m <<= 1) x += __shfl_xor(x, m, 32);
  return x;
}

// ---------------- weight prep ----------------
// Wk/Wv: [H][C][hd] f32  ->  Wt: [n=h*hd+d][k=C] bf16 (B^T layout)
__global__ __launch_bounds__(256) void conv_head_weight(
    const float* __restrict__ W, __bf16* __restrict__ Wt, int C, int hd, int n_elem) {
  int o = blockIdx.x * blockDim.x + threadIdx.x;
  if (o >= n_elem) return;
  int n = o / C, k = o % C;
  int h = n / hd, d = n % hd;
  Wt[o] = (__bf16)W[(size_t)h * C * hd + (size_t)k * hd + d];
}

__global__ __launch_bounds__(256) void f2bf_kernel(
    const float* __restrict__ W, __bf16* __restrict__ O, int n) {
  int i = blockIdx.x * blockDim.x + threadIdx.x;
  if (i < n) O[i] = (__bf16)W[i];
}

// ---------------- LayerNorm over the T axis (dim=1), unbiased variance ----------------
__global__ __launch_bounds__(256) void ln_seq_kernel(
    const float* __restrict__ x, const float* __restrict__ g,
    const float* __restrict__ beta, __bf16* __restrict__ out, int BC, int T, int C) {
  int id = blockIdx.x * blockDim.x + threadIdx.x;  // over B*C columns
  if (id >= BC) return;
  int b = id / C, c = id % C;
  const float* col = x + (size_t)b * T * C + c;
  float s = 0.f, ss = 0.f;
  for (int t = 0; t < T; ++t) { float v = col[(size_t)t * C]; s += v; ss += v * v; }
  float mean = s / (float)T;
  float var = (ss - s * mean) / (float)(T - 1);
  float rstd = rsqrtf(var + 1e-5f);
  float gc = g[c], bc = beta[c];
  __bf16* o = out + (size_t)b * T * C + c;
  for (int t = 0; t < T; ++t)
    o[(size_t)t * C] = (__bf16)(gc * (col[(size_t)t * C] - mean) * rstd + bc);
}

// ---------------- generic WMMA GEMM: D[M,N] = A[M,K] @ Bt[N,K]^T (+bias,+res,ReLU) ----
// 256 threads = 8 waves; block tile 128x128, BK=64 (2 k-steps per stage),
// double-buffered LDS filled by async global->LDS DMA; wave tile 32x64 (2x4 WMMA tiles)
template <bool OUT_BF, bool RELU, bool BIAS, bool RES>
__global__ __launch_bounds__(256) void gemm_bf16_wmma(
    const __bf16* __restrict__ A, const __bf16* __restrict__ Bt,
    const float* __restrict__ bias, const float* __restrict__ res,
    float* __restrict__ outF, __bf16* __restrict__ outB, int M, int N, int K) {
  __shared__ __bf16 sA[2][128][72];
  __shared__ __bf16 sB[2][128][72];
  const int tid = threadIdx.x;
  const int bm = blockIdx.y * 128, bn = blockIdx.x * 128;
  const int w = tid >> 5, lane = tid & 31, hi = lane >> 4, ln = lane & 15;
  const int wm = (w >> 1) * 32, wn = (w & 1) * 64;

  const v8f vzero = {0, 0, 0, 0, 0, 0, 0, 0};
  v8f acc[2][4];
  for (int i = 0; i < 2; ++i)
    for (int j = 0; j < 4; ++j) acc[i][j] = vzero;

  // stage issue: 128 rows x 64 cols per tile; 8-elem (16B) chunks; 4 chunks/thread/tile
  auto issue_stage = [&](int k0, int buf) {
    for (int it = 0; it < 4; ++it) {
      const int cid = tid + it * 256;          // 0..1023
      const int r = cid >> 3, cg = (cid & 7) * 8;
      async_copy_b128(&A[(size_t)(bm + r) * K + k0 + cg], &sA[buf][r][cg]);
      async_copy_b128(&Bt[(size_t)(bn + r) * K + k0 + cg], &sB[buf][r][cg]);
    }
  };

  const int nstages = K >> 6;
  issue_stage(0, 0);
  for (int s = 0; s < nstages; ++s) {
    const int buf = s & 1;
    wait_async0();        // my stage-s loads landed in LDS
    __syncthreads();      // everyone's stage-s loads landed; stage s-1 compute done
    if (s + 1 < nstages) issue_stage((s + 1) << 6, buf ^ 1);
    for (int ks = 0; ks < 2; ++ks) {
      v16bf af[2], bfr[4];
      for (int mt = 0; mt < 2; ++mt)
        af[mt] = ld_fragA(&sA[buf][wm + mt * 16 + ln][ks * 32], hi);
      for (int nt = 0; nt < 4; ++nt)
        bfr[nt] = ld_fragB(&sB[buf][wn + nt * 16 + ln][ks * 32], hi);
      for (int mt = 0; mt < 2; ++mt)
        for (int nt = 0; nt < 4; ++nt)
          acc[mt][nt] = WMMA_BF16(af[mt], bfr[nt], acc[mt][nt]);
    }
  }

  for (int mt = 0; mt < 2; ++mt)
    for (int nt = 0; nt < 4; ++nt) {
      const int n = bn + wn + nt * 16 + ln;
      for (int v = 0; v < 8; ++v) {
        const int m = bm + wm + mt * 16 + v + 8 * hi;  // C/D layout: VGPR v -> row v+8*hi
        float val = acc[mt][nt][v];
        if (BIAS) val += bias[n];
        if (RES) val += res[(size_t)m * N + n];
        if (RELU) val = fmaxf(val, 0.0f);
        if (OUT_BF) outB[(size_t)m * N + n] = (__bf16)val;
        else        outF[(size_t)m * N + n] = val;
      }
    }
}

// ---------------- flash attention (q == k, causal, hd = 64) ----------------
// grid: (T/128, H, B); 256 threads = 8 waves, wave w owns rows [w*16, w*16+16)
__global__ __launch_bounds__(256) void attn_kernel(
    const __bf16* __restrict__ qk, const __bf16* __restrict__ vv,
    __bf16* __restrict__ out, int T, int C) {
  __shared__ __bf16 sK[128][72];       // K tile [s][k]
  __shared__ __bf16 sVt[64][136];      // V tile transposed [d][s]
  __shared__ __bf16 sP[8][16][136];    // per-wave P strip [row][s]

  const int it = blockIdx.x, h = blockIdx.y, b = blockIdx.z;
  const int tid = threadIdx.x;
  const int w = tid >> 5, lane = tid & 31, hi = lane >> 4, ln = lane & 15;
  const __bf16* qbase = qk + ((size_t)b * T) * C + (size_t)h * 64;
  const __bf16* vbase = vv + ((size_t)b * T) * C + (size_t)h * 64;

  // Q fragments (A-layout): lane holds row it*128 + w*16 + ln
  const __bf16* qrow = qbase + (size_t)(it * 128 + w * 16 + ln) * C;
  v16bf qf0 = ld_fragA(qrow, hi);
  v16bf qf1 = ld_fragA(qrow + 32, hi);

  const v8f vzero = {0, 0, 0, 0, 0, 0, 0, 0};
  float mrow[8], lrow[8];
  v8f o_acc[4];
  for (int v = 0; v < 8; ++v) { mrow[v] = -3.0e38f; lrow[v] = 0.f; }
  for (int dt = 0; dt < 4; ++dt) o_acc[dt] = vzero;

  const float scale = 0.125f;  // hd^-0.5, hd = 64

  for (int j = 0; j <= it; ++j) {
    // cooperative load of K tile and transposed V tile
    for (int e = tid; e < 1024; e += 256) {
      const int r = e >> 3, cg = (e & 7) * 8;
      *(v8bf*)&sK[r][cg] = *(const v8bf*)(qbase + (size_t)(j * 128 + r) * C + cg);
      v8bf vt = *(const v8bf*)(vbase + (size_t)(j * 128 + r) * C + cg);
      for (int q = 0; q < 8; ++q) sVt[cg + q][r] = vt[q];
    }
    // prefetch next j tile into cache (global_prefetch_b8)
    if (j < it) {
      __builtin_prefetch(qbase + (size_t)((j + 1) * 128 + (tid >> 1)) * C + (tid & 1) * 32, 0, 1);
      __builtin_prefetch(vbase + (size_t)((j + 1) * 128 + (tid >> 1)) * C + (tid & 1) * 32, 0, 1);
    }
    __syncthreads();

    // S strip: 16 rows x 128 cols via 16 WMMAs
    v8f sacc[8];
    for (int nt = 0; nt < 8; ++nt) {
      const __bf16* krow = &sK[nt * 16 + ln][0];
      v8f z = vzero;
      z = WMMA_BF16(qf0, ld_fragB(krow, hi), z);
      sacc[nt] = WMMA_BF16(qf1, ld_fragB(krow + 32, hi), z);
    }

    // scale + causal mask + row max
    float rmax[8];
    for (int v = 0; v < 8; ++v) rmax[v] = -3.0e38f;
    const bool diag = (j == it);
    for (int nt = 0; nt < 8; ++nt)
      for (int v = 0; v < 8; ++v) {
        float sv = sacc[nt][v] * scale;
        if (diag && (nt * 16 + ln) > (w * 16 + v + 8 * hi)) sv = -1.0e30f;
        sacc[nt][v] = sv;
        rmax[v] = fmaxf(rmax[v], sv);
      }
    for (int v = 0; v < 8; ++v) rmax[v] = hmax16(rmax[v]);

    // online softmax update
    float alpha[8], rsum[8];
    for (int v = 0; v < 8; ++v) {
      float mn = fmaxf(mrow[v], rmax[v]);
      alpha[v] = __expf(mrow[v] - mn);
      mrow[v] = mn;
      rsum[v] = 0.f;
    }
    for (int nt = 0; nt < 8; ++nt)
      for (int v = 0; v < 8; ++v) {
        float p = __expf(sacc[nt][v] - mrow[v]);
        rsum[v] += p;
        sP[w][v + 8 * hi][nt * 16 + ln] = (__bf16)p;  // C/D -> memory layout
      }
    for (int v = 0; v < 8; ++v) {
      rsum[v] = hsum16(rsum[v]);
      lrow[v] = lrow[v] * alpha[v] + rsum[v];
    }
    for (int dt = 0; dt < 4; ++dt)
      for (int v = 0; v < 8; ++v) o_acc[dt][v] *= alpha[v];

    // O += P @ V  (per-wave sP: in-order LDS within wave, no barrier needed)
    for (int kk = 0; kk < 4; ++kk) {
      v16bf pf = ld_fragA(&sP[w][ln][kk * 32], hi);
      for (int dt = 0; dt < 4; ++dt) {
        v16bf vb = ld_fragB(&sVt[dt * 16 + ln][kk * 32], hi);
        o_acc[dt] = WMMA_BF16(pf, vb, o_acc[dt]);
      }
    }
    __syncthreads();
  }

  // normalize and store [b, t, h*64 + d] as bf16 for the projection GEMM
  for (int dt = 0; dt < 4; ++dt) {
    const int d = dt * 16 + ln;
    for (int v = 0; v < 8; ++v) {
      const int t = it * 128 + w * 16 + v + 8 * hi;
      out[((size_t)b * T + t) * C + h * 64 + d] = (__bf16)(o_acc[dt][v] / lrow[v]);
    }
  }
}

// ---------------- launch ----------------
extern "C" void kernel_launch(void* const* d_in, const int* in_sizes, int n_in,
                              void* d_out, int out_size, void* d_ws, size_t ws_size,
                              hipStream_t stream) {
  const int T = 1024, C = 1024, H = 16, hd = 64, C4 = 4096;
  const int B = in_sizes[0] / (T * C);
  const int M = B * T;

  const float* x   = (const float*)d_in[0];
  const float* Wk  = (const float*)d_in[1];
  const float* Wv  = (const float*)d_in[2];
  const float* Wp  = (const float*)d_in[3];
  const float* bp  = (const float*)d_in[4];
  const float* W1  = (const float*)d_in[5];
  const float* b1  = (const float*)d_in[6];
  const float* W2  = (const float*)d_in[7];
  const float* b2  = (const float*)d_in[8];
  const float* g1  = (const float*)d_in[9];
  const float* be1 = (const float*)d_in[10];
  const float* g2  = (const float*)d_in[11];
  const float* be2 = (const float*)d_in[12];
  float* out = (float*)d_out;

  char* ws = (char*)d_ws;
  size_t off = 0;
  auto alloc = [&](size_t bytes) {
    char* p = ws + off;
    off += (bytes + 255) & ~(size_t)255;
    return p;
  };
  __bf16* h_bf    = (__bf16*)alloc((size_t)M * C * 2);
  __bf16* qk_bf   = (__bf16*)alloc((size_t)M * C * 2);
  __bf16* v_bf    = (__bf16*)alloc((size_t)M * C * 2);
  __bf16* attn_bf = (__bf16*)alloc((size_t)M * C * 2);
  float*  x1      = (float*) alloc((size_t)M * C * 4);
  __bf16* h2_bf   = (__bf16*)alloc((size_t)M * C * 2);
  __bf16* ff1_bf  = (__bf16*)alloc((size_t)M * C4 * 2);
  __bf16* WkT     = (__bf16*)alloc((size_t)C * C * 2);
  __bf16* WvT     = (__bf16*)alloc((size_t)C * C * 2);
  __bf16* Wpbf    = (__bf16*)alloc((size_t)C * C * 2);
  __bf16* W1bf    = (__bf16*)alloc((size_t)C4 * C * 2);
  __bf16* W2bf    = (__bf16*)alloc((size_t)C * C4 * 2);
  (void)ws_size; (void)n_in; (void)out_size;

  dim3 blk(256);

  // weight prep (bf16, B^T layouts)
  conv_head_weight<<<(C * C) / 256, blk, 0, stream>>>(Wk, WkT, C, hd, C * C);
  conv_head_weight<<<(C * C) / 256, blk, 0, stream>>>(Wv, WvT, C, hd, C * C);
  f2bf_kernel<<<(C * C) / 256, blk, 0, stream>>>(Wp, Wpbf, C * C);
  f2bf_kernel<<<(C4 * C) / 256, blk, 0, stream>>>(W1, W1bf, C4 * C);
  f2bf_kernel<<<(C * C4) / 256, blk, 0, stream>>>(W2, W2bf, C * C4);

  // LN1 -> h
  ln_seq_kernel<<<(B * C + 255) / 256, blk, 0, stream>>>(x, g1, be1, h_bf, B * C, T, C);

  // qk = h @ Wk, v = h @ Wv   (bf16 out)
  gemm_bf16_wmma<true, false, false, false><<<dim3(C / 128, M / 128), blk, 0, stream>>>(
      h_bf, WkT, nullptr, nullptr, nullptr, qk_bf, M, C, C);
  gemm_bf16_wmma<true, false, false, false><<<dim3(C / 128, M / 128), blk, 0, stream>>>(
      h_bf, WvT, nullptr, nullptr, nullptr, v_bf, M, C, C);

  // causal attention (q == k)
  attn_kernel<<<dim3(T / 128, H, B), blk, 0, stream>>>(qk_bf, v_bf, attn_bf, T, C);

  // x1 = x + attn @ Wp^T + bp   (f32 out)
  gemm_bf16_wmma<false, false, true, true><<<dim3(C / 128, M / 128), blk, 0, stream>>>(
      attn_bf, Wpbf, bp, x, x1, nullptr, M, C, C);

  // LN2 -> h2
  ln_seq_kernel<<<(B * C + 255) / 256, blk, 0, stream>>>(x1, g2, be2, h2_bf, B * C, T, C);

  // ff1 = relu(h2 @ W1^T + b1)  (bf16 out)
  gemm_bf16_wmma<true, true, true, false><<<dim3(C4 / 128, M / 128), blk, 0, stream>>>(
      h2_bf, W1bf, b1, nullptr, nullptr, ff1_bf, M, C4, C);

  // out = x1 + ff1 @ W2^T + b2  (f32 out)
  gemm_bf16_wmma<false, false, true, true><<<dim3(C / 128, M / 128), blk, 0, stream>>>(
      ff1_bf, W2bf, b2, x1, out, nullptr, M, C, C4);
}
